// BiLSTM_44693429682204
// MI455X (gfx1250) — matrix operand order repaired and checked by reading, not compiled, not run on hardware
//
#include <hip/hip_runtime.h>
#include <hip/hip_bf16.h>

// ---------------------------------------------------------------------------
// BiLSTM for MI455X (gfx1250): B=128, T=512, I=512, H=512.
//   Kernel 1: f32 -> bf16 conversion of x and weights (grid-stride).
//   Kernel 2: gx = x @ W_ih^T via v_wmma_f32_16x16x32_bf16, full-GPU grid.
//   Kernel 3: persistent scan. 16 WGs = 8 batch tiles x 2 dirs, 32 waves each.
//             Per step each wave: 16 A-frag LDS loads, 64 B-frag L2 loads,
//             64 WMMAs (4 gate tiles x K=512/32), then LSTM cell math in regs.
// ---------------------------------------------------------------------------

typedef __attribute__((ext_vector_type(16))) __bf16 v16bf;
typedef __attribute__((ext_vector_type(8)))  float  v8f;

#define B_ 128
#define T_ 512
#define I_ 512
#define H_ 512
#define G_ 2048  // 4*H

__device__ __forceinline__ unsigned short f2bf(float f) {
  unsigned int u = __float_as_uint(f);
  u += 0x7fffu + ((u >> 16) & 1u);            // round-to-nearest-even
  return (unsigned short)(u >> 16);
}
__device__ __forceinline__ float bf2f(unsigned short h) {
  return __uint_as_float(((unsigned int)h) << 16);
}
__device__ __forceinline__ float sigm(float x) { return 1.0f / (1.0f + __expf(-x)); }
__device__ __forceinline__ float tanhx(float x) { return 1.0f - 2.0f / (1.0f + __expf(2.0f * x)); }

// 16-bit WMMA fragment = 8 VGPRs; memory image is two 16-byte chunks.
union Frag {
  v16bf v;
  uint4 q[2];
};

// ---------------------------------------------------------------------------
__global__ void bilstm_cvt_bf16(const float* __restrict__ src,
                                unsigned short* __restrict__ dst, int n) {
  int i = blockIdx.x * blockDim.x + threadIdx.x;
  int stride = gridDim.x * blockDim.x;
  for (; i < n; i += stride) dst[i] = f2bf(src[i]);
}

// ---------------------------------------------------------------------------
// gx[dir][t][b][g] = sum_i x[b,t,i] * W_ih[dir][g,i]   (bias added in scan)
// A rows are (b*T + t); T%16==0 so a 16-row M tile stays within one b.
__global__ __launch_bounds__(256) void bilstm_gx_gemm(
    const unsigned short* __restrict__ xb,   // [B*T, I] bf16
    const unsigned short* __restrict__ wih,  // [2][G, I] bf16 (row-major = B^T layout)
    unsigned short* __restrict__ gx) {       // [2][T][B][G] bf16
  const int lane = threadIdx.x & 31;
  const int wave = threadIdx.x >> 5;
  const int hi   = lane >> 4;
  const int l15  = lane & 15;
  const int dir  = blockIdx.z;
  const int R0   = blockIdx.x * 16;          // base row in (b*T + t)
  const int b    = R0 >> 9;                  // T = 512
  const int t0   = R0 & (T_ - 1);
  const int n    = (blockIdx.y * 8 + wave) * 16 + l15;

  const unsigned short* ap = xb + (size_t)(R0 + l15) * I_;
  const unsigned short* wp = wih + (size_t)dir * ((size_t)G_ * I_) + (size_t)n * I_;

  v8f acc = {0.f, 0.f, 0.f, 0.f, 0.f, 0.f, 0.f, 0.f};
  for (int kt = 0; kt < I_ / 32; ++kt) {
    Frag a, w;
    // A 16x32 bf16: lane half 'hi' holds K = hi*8..hi*8+7 and 16+hi*8..
    a.q[0] = *(const uint4*)(ap + kt * 32 + hi * 8);
    a.q[1] = *(const uint4*)(ap + kt * 32 + 16 + hi * 8);
    // B: lane holds column n, same K order as A (hi*8.. and 16+hi*8..).
    w.q[0] = *(const uint4*)(wp + kt * 32 + hi * 8);
    w.q[1] = *(const uint4*)(wp + kt * 32 + 16 + hi * 8);
    acc = __builtin_amdgcn_wmma_f32_16x16x32_bf16(false, a.v, false, w.v,
                                                  (short)0, acc, false, false);
  }
#pragma unroll
  for (int r = 0; r < 8; ++r) {
    const int t = t0 + r + 8 * hi;           // C layout: M = r + 8*hi, N = l15
    gx[(((size_t)dir * T_ + t) * B_ + b) * G_ + n] = f2bf(acc[r]);
  }
}

// ---------------------------------------------------------------------------
// Persistent recurrence. grid = (8 batch tiles, 2 dirs), block = 1024 (32 waves).
// Wave j owns hidden columns [j*16, j*16+16) for all four gates.
__global__ __launch_bounds__(1024) void bilstm_scan(
    const unsigned short* __restrict__ gx,    // [2][T][B][G] bf16
    const unsigned short* __restrict__ whh,   // [2][G, H] bf16
    const float* __restrict__ bih_f, const float* __restrict__ bhh_f,
    const float* __restrict__ bih_b, const float* __restrict__ bhh_b,
    const unsigned char* __restrict__ mask,   // [B, T]
    float* __restrict__ out) {                // [B, T, 2H]
  __shared__ v16bf hbuf[16][33];              // 16 x 512 bf16 h state, +32B row pad
  unsigned short* hs = (unsigned short*)hbuf;
  const int ROWS = 33 * 16;                   // ushorts per padded row (528)

  const int lane = threadIdx.x & 31;
  const int wave = threadIdx.x >> 5;
  const int hi   = lane >> 4;
  const int l15  = lane & 15;
  const int j    = wave;                      // hidden col tile 0..31
  const int b0   = blockIdx.x * 16;
  const int dir  = blockIdx.y;

  const float* bih = dir ? bih_b : bih_f;
  const float* bhh = dir ? bhh_b : bhh_f;
  const unsigned short* wp = whh + (size_t)dir * ((size_t)G_ * H_);

  float bias[4];
#pragma unroll
  for (int g = 0; g < 4; ++g)
    bias[g] = bih[g * H_ + j * 16 + l15] + bhh[g * H_ + j * 16 + l15];

  for (int i = threadIdx.x; i < 16 * ROWS; i += blockDim.x) hs[i] = 0;
  float c8[8];
#pragma unroll
  for (int r = 0; r < 8; ++r) c8[r] = 0.f;
  __syncthreads();

  const v8f vzero = {0.f, 0.f, 0.f, 0.f, 0.f, 0.f, 0.f, 0.f};

  for (int step = 0; step < T_; ++step) {
    const int t = dir ? (T_ - 1 - step) : step;

    // Input projections + masks (no dependency on h): issue early.
    float gxv[4][8];
    const unsigned short* gxt = gx + ((size_t)dir * T_ + t) * ((size_t)B_ * G_);
#pragma unroll
    for (int g = 0; g < 4; ++g)
#pragma unroll
      for (int r = 0; r < 8; ++r)
        gxv[g][r] = bf2f(gxt[(size_t)(b0 + 8 * hi + r) * G_ + g * H_ + j * 16 + l15]);
    unsigned char mk[8];
#pragma unroll
    for (int r = 0; r < 8; ++r) mk[r] = mask[(size_t)(b0 + 8 * hi + r) * T_ + t];
    if (step + 1 < T_) {                      // pull next step's gx toward L2/L1
      const int tn = dir ? (T_ - 2 - step) : (step + 1);
      __builtin_prefetch(gx + (((size_t)dir * T_ + tn) * B_ + b0 + 8 * hi) * G_ +
                             j * 16 + l15, 0, 1);
    }

    // gates = h @ W_hh^T (four 16x16 tiles, K = 512)
    v8f acc[4] = {vzero, vzero, vzero, vzero};
    const unsigned short* hrow = hs + (size_t)l15 * ROWS;
    for (int kt = 0; kt < H_ / 32; ++kt) {
      Frag a;
      a.q[0] = *(const uint4*)(hrow + kt * 32 + hi * 8);
      a.q[1] = *(const uint4*)(hrow + kt * 32 + 16 + hi * 8);
#pragma unroll
      for (int g = 0; g < 4; ++g) {
        Frag w;
        const unsigned short* bp =
            wp + (size_t)(g * H_ + j * 16 + l15) * H_ + kt * 32 + hi * 8;
        w.q[0] = *(const uint4*)(bp);
        w.q[1] = *(const uint4*)(bp + 16);
        acc[g] = __builtin_amdgcn_wmma_f32_16x16x32_bf16(false, a.v, false, w.v,
                                                         (short)0, acc[g], false, false);
      }
    }
    __syncthreads();                          // all waves done reading h

    // LSTM cell math; i/f/g/o share (m, n) positions across acc[0..3].
#pragma unroll
    for (int r = 0; r < 8; ++r) {
      const float iv = acc[0][r] + gxv[0][r] + bias[0];
      const float fv = acc[1][r] + gxv[1][r] + bias[1];
      const float gv = acc[2][r] + gxv[2][r] + bias[2];
      const float ov = acc[3][r] + gxv[3][r] + bias[3];
      const float cn = sigm(fv) * c8[r] + sigm(iv) * tanhx(gv);
      const float hn = sigm(ov) * tanhx(cn);
      const int m = 8 * hi + r;
      const bool on = mk[r] != 0;
      if (on) {
        c8[r] = cn;
        hs[(size_t)m * ROWS + j * 16 + l15] = f2bf(hn);
      }
      out[((size_t)(b0 + m) * T_ + t) * (2 * H_) + dir * H_ + j * 16 + l15] =
          on ? hn : 0.f;
    }
    __syncthreads();                          // h update visible before next step
  }
}

// ---------------------------------------------------------------------------
extern "C" void kernel_launch(void* const* d_in, const int* in_sizes, int n_in,
                              void* d_out, int out_size, void* d_ws, size_t ws_size,
                              hipStream_t stream) {
  (void)in_sizes; (void)n_in; (void)out_size; (void)ws_size;
  const float* x            = (const float*)d_in[0];
  const unsigned char* mask = (const unsigned char*)d_in[1];
  const float* Wih_f = (const float*)d_in[2];
  const float* Whh_f = (const float*)d_in[3];
  const float* bih_f = (const float*)d_in[4];
  const float* bhh_f = (const float*)d_in[5];
  const float* Wih_b = (const float*)d_in[6];
  const float* Whh_b = (const float*)d_in[7];
  const float* bih_b = (const float*)d_in[8];
  const float* bhh_b = (const float*)d_in[9];
  float* out = (float*)d_out;

  // Workspace layout (bf16 elements): x | W_ih[2] | W_hh[2] | gx  (~612 MB)
  unsigned short* ws = (unsigned short*)d_ws;
  const size_t XB = (size_t)B_ * T_ * I_;     // 33,554,432
  const size_t WE = (size_t)G_ * I_;          // 1,048,576
  unsigned short* xb  = ws;
  unsigned short* wih = xb + XB;
  unsigned short* whh = wih + 2 * WE;
  unsigned short* gxb = whh + 2 * WE;

  bilstm_cvt_bf16<<<4096, 256, 0, stream>>>(x, xb, (int)XB);
  bilstm_cvt_bf16<<<1024, 256, 0, stream>>>(Wih_f, wih,      (int)WE);
  bilstm_cvt_bf16<<<1024, 256, 0, stream>>>(Wih_b, wih + WE, (int)WE);
  bilstm_cvt_bf16<<<1024, 256, 0, stream>>>(Whh_f, whh,      (int)WE);
  bilstm_cvt_bf16<<<1024, 256, 0, stream>>>(Whh_b, whh + WE, (int)WE);

  dim3 ggrid(B_ * T_ / 16, (G_ / 16) / 8, 2); // (4096, 16, 2)
  bilstm_gx_gemm<<<ggrid, 256, 0, stream>>>(xb, wih, gxb);

  bilstm_scan<<<dim3(B_ / 16, 2), 1024, 0, stream>>>(
      gxb, whh, bih_f, bhh_f, bih_b, bhh_b, mask, out);
}
